// CoordinatesToSequenceIPA_29583734735217
// MI455X (gfx1250) — matrix operand order repaired and checked by reading, not compiled, not run on hardware
//
#include <hip/hip_runtime.h>
#include <hip/hip_bf16.h>
#include <math.h>

typedef __attribute__((ext_vector_type(16))) __bf16 v16bf;
typedef __attribute__((ext_vector_type(8)))  float  v8f;
typedef __attribute__((ext_vector_type(2)))  float  v2f;
typedef unsigned short u16;
typedef unsigned int   u32;
typedef unsigned long long u64;

#define DI __device__ __forceinline__

// ---------------- problem constants ----------------
static constexpr int kBn = 2, kNn = 512, kCS = 512, kCZ = 128;
static constexpr int kH = 8, kC = 16, kPQ = 8, kPV = 12;
static constexpr int kCat = kH*kC + kH*kPV*4 + kH*kCZ;   // 1536
static constexpr float kWL  = 0.14433756729740643f;      // sqrt(1/(3*C))
static constexpr float kS3  = 0.5773502691896258f;       // sqrt(1/3)
static constexpr float kWC  = 0.09622504486493763f;      // sqrt(1/(3*PQ*9/2))
static constexpr float kINF = 1e5f;
static constexpr float kEPS = 1e-8f;

// ---------------- helpers ----------------
// native f32 -> bf16 convert (backend emits v_cvt_pk_bf16_f32 on gfx1250)
DI __bf16 bf(float f)   { return (__bf16)f; }
DI u16    bfbits(float f){ return __builtin_bit_cast(u16, (__bf16)f); }
DI __bf16 bfu(u16 s)    { return __builtin_bit_cast(__bf16, s); }

DI v8f wmma_bf16(v16bf a, v16bf b, v8f c) {
  return __builtin_amdgcn_wmma_f32_16x16x32_bf16(false, a, false, b, (short)0, c, false, false);
}

#if defined(__gfx1250__) && __has_builtin(__builtin_amdgcn_wmma_f32_16x16x4_f32)
#define HAVE_WMMA_F32X4 1
DI v8f wmma_f32x4(v2f a, v2f b, v8f c) {
  return __builtin_amdgcn_wmma_f32_16x16x4_f32(false, a, false, b, (short)0, c, false, false);
}
#endif

// Async global -> LDS staging (CDNA5): 16B per issue, tracked by ASYNCcnt.
#if defined(__gfx1250__)
#define K_ASYNC 1
#endif

DI void stage_async(const float* g, float* l) {
#ifdef K_ASYNC
  u32 lo = (u32)(u64)(size_t)l;          // low 32 bits of LDS flat address
  u64 ga = (u64)(size_t)g;
  asm volatile("global_load_async_to_lds_b128 %0, %1, off" :: "v"(lo), "v"(ga) : "memory");
#else
  float4 v = *(const float4*)g;
  *(float4*)l = v;
#endif
}
DI void wait_async() {
#ifdef K_ASYNC
  asm volatile("s_wait_asynccnt 0x0" ::: "memory");
#endif
}

// ==========================================================================
// K1: generic projection GEMM  out[M,Nw] = A[M,K] @ W[K,Nw] + bias
// 1 wave per 16x16 tile; bf16 WMMA, K stepped by 32.
// ==========================================================================
__global__ __launch_bounds__(32)
void k_proj(const float* __restrict__ A, const float* __restrict__ W,
            const float* __restrict__ bias, float* __restrict__ out,
            int K, int Nw) {
  const int mt = blockIdx.x, nt = blockIdx.y;
  const int lane = threadIdx.x;
  const int r = lane & 15, half = lane >> 4;
  const int kb = half ? 8 : 0, ks = half ? 16 : 0;
  const float* Ar = A + (size_t)(mt*16 + r) * K;
  const int col = nt*16 + r;
  v8f acc = {};
  for (int k0 = 0; k0 < K; k0 += 32) {
    v16bf a, bm;
#pragma unroll
    for (int i2 = 0; i2 < 8; ++i2) {
      a[i2]     = bf(Ar[k0 + kb + i2]);
      a[8 + i2] = bf(Ar[k0 + 16 + kb + i2]);
    }
#pragma unroll
    for (int m = 0; m < 16; ++m)
      bm[m] = bf(W[(size_t)(k0 + kb + m + (m & 8)) * Nw + col]);
    acc = wmma_bf16(a, bm, acc);
  }
  const float bv = bias[col];
  const int rbase = half * 8;
#pragma unroll
  for (int rr = 0; rr < 8; ++rr)
    out[(size_t)(mt*16 + rbase + rr) * Nw + col] = acc[rr] + bv;
}

// ==========================================================================
// K2: per-(b,h,n) squared point norms (q_pts / k_pts include +t)
// ==========================================================================
__global__ void k_sq(const float* __restrict__ qp, const float* __restrict__ kvp,
                     const float* __restrict__ x_t, const float* __restrict__ mask,
                     float* __restrict__ sqq, float* __restrict__ sqk) {
  int idx = blockIdx.x * blockDim.x + threadIdx.x;
  if (idx >= kBn*kH*kNn) return;
  int n = idx % kNn, h = (idx / kNn) % kH, b = idx / (kNn*kH);
  float m = mask[b*kNn + n];
  float t0 = x_t[(b*kNn+n)*3+0]*m, t1 = x_t[(b*kNn+n)*3+1]*m, t2 = x_t[(b*kNn+n)*3+2]*m;
  const float* qb = qp + (size_t)(b*kNn+n) * (kH*kPQ*3);
  float s = 0.f;
  for (int p = 0; p < kPQ; ++p) {
    float vx = qb[(0*kH+h)*kPQ + p] + t0;
    float vy = qb[(1*kH+h)*kPQ + p] + t1;
    float vz = qb[(2*kH+h)*kPQ + p] + t2;
    s += vx*vx + vy*vy + vz*vz;
  }
  sqq[(b*kH + h)*kNn + n] = s;
  const float* kb2 = kvp + (size_t)(b*kNn+n) * (kH*(kPQ+kPV)*3);
  float s2 = 0.f;
  for (int p = 0; p < kPQ; ++p) {
    float vx = kb2[(0*kH+h)*(kPQ+kPV) + p] + t0;
    float vy = kb2[(1*kH+h)*(kPQ+kPV) + p] + t1;
    float vz = kb2[(2*kH+h)*(kPQ+kPV) + p] + t2;
    s2 += vx*vx + vy*vy + vz*vz;
  }
  sqk[(b*kH + h)*kNn + n] = s2;
}

// ==========================================================================
// K3: bias[b,i,j,h] = pair_rep[b,i,j,:] @ Wb + bb   (WMMA, H padded to 16)
// 128 threads / block; 64 flattened (b,i,j) rows staged to LDS via async DMA,
// 4 waves each consume a 16-row tile.
// ==========================================================================
__global__ __launch_bounds__(128)
void k_bias(const float* __restrict__ pr, const float* __restrict__ Wb,
            const float* __restrict__ bb, float* __restrict__ biasw) {
  const int blk = blockIdx.x;                        // 64 rows per block
  const int tid = threadIdx.x, lane = tid & 31, wave = tid >> 5;
  const int r = lane & 15, half = lane >> 4;
  const int kb = half ? 8 : 0, ks = half ? 16 : 0;

  __shared__ float Al[64 * kCZ];                     // 32 KB staged pair_rep

  const float* g = pr + (size_t)blk * 64 * kCZ;
#pragma unroll
  for (int rr = 0; rr < 16; ++rr) {                  // 2048 float4 / 128 thr
    int idx4 = rr * 128 + tid;
    stage_async(g + idx4*4, &Al[idx4*4]);
  }
  wait_async();
  __syncthreads();

  const float* Ar = Al + (wave*16 + r) * kCZ;
  v8f acc = {};
  for (int k0 = 0; k0 < kCZ; k0 += 32) {
    v16bf a, bm;
#pragma unroll
    for (int i2 = 0; i2 < 8; ++i2) {
      a[i2]     = bf(Ar[k0 + kb + i2]);
      a[8 + i2] = bf(Ar[k0 + 16 + kb + i2]);
    }
#pragma unroll
    for (int m = 0; m < 16; ++m)
      bm[m] = (r < kH) ? bf(Wb[(k0 + kb + m + (m & 8)) * kH + r]) : bfu(0);
    acc = wmma_bf16(a, bm, acc);
  }
  if (r < kH) {
    const float bv = bb[r];
    const int rbase = half * 8;
#pragma unroll
    for (int rr = 0; rr < 8; ++rr)
      biasw[(size_t)(blk*64 + wave*16 + rbase + rr) * kH + r] = acc[rr] + bv;
  }
}

// ==========================================================================
// K4: fused attention per (b, h, i-tile of 16): logits (f32 WMMA if available)
//     -> softmax (LDS) -> P (bf16) -> o = P@v, o_pt = P@v_pts (bf16 WMMA)
// Block = 128 threads = 4 waves.
// ==========================================================================
__global__ __launch_bounds__(128)
void k_attn(const float* __restrict__ q, const float* __restrict__ kv,
            const float* __restrict__ qp, const float* __restrict__ kvp,
            const float* __restrict__ x_t, const float* __restrict__ mask,
            const float* __restrict__ biasw, const float* __restrict__ sqq,
            const float* __restrict__ sqk, const float* __restrict__ head_w,
            u16* __restrict__ Pg, float* __restrict__ feats) {
  const int it = blockIdx.x, h = blockIdx.y, b = blockIdx.z;
  const int i0 = it * 16;
  const int tid = threadIdx.x, lane = tid & 31, wave = tid >> 5;
  const int r = lane & 15, half = lane >> 4, rbase = half * 8;
  const int kb = half ? 8 : 0, ks = half ? 16 : 0;

  __shared__ float S[16][kNn];          // logits -> probs (f32)
  __shared__ u16   Pl[16 * kNn];        // probs bf16 for phase-2 A fragments
  __shared__ float Opt[16][48];         // o_pt staging (36 used)
  __shared__ float sq16[16], mI[16], tI[16][3];
  __shared__ float red[16][8], rowst[16];

  const float hw = logf(1.f + expf(head_w[h])) * kWC;

  if (tid < 16) {
    int i = i0 + tid;
    float m = mask[b*kNn + i];
    mI[tid] = m;
    sq16[tid] = sqq[(b*kH + h)*kNn + i];
#pragma unroll
    for (int d = 0; d < 3; ++d) tI[tid][d] = x_t[(b*kNn+i)*3 + d] * m;
  }
  __syncthreads();

#ifdef HAVE_WMMA_F32X4
  // ------- A operands in full f32 for V_WMMA_F32_16X16X4_F32 -------
  // per step s the lane supplies A[row, 4s + 2*half + {0,1}]
  float aqf[8], axf[12];
  {
    const float* qr = q + (size_t)(b*kNn + i0 + r) * (kH*kC) + h*kC;
#pragma unroll
    for (int s = 0; s < 4; ++s) {
      aqf[2*s+0] = qr[4*s + 2*half + 0];
      aqf[2*s+1] = qr[4*s + 2*half + 1];
    }
    const float* qpr = qp + (size_t)(b*kNn + i0 + r) * (kH*kPQ*3);
#pragma unroll
    for (int s = 0; s < 6; ++s) {
#pragma unroll
      for (int t = 0; t < 2; ++t) {
        int K = 4*s + 2*half + t;       // < 24 always
        int p = K / 3, x = K - 3*p;
        axf[2*s+t] = qpr[(x*kH + h)*kPQ + p] + tI[r][x];
      }
    }
  }
#else
  v16bf aq, ax;
  {
    const float* qr = q + (size_t)(b*kNn + i0 + r) * (kH*kC) + h*kC;
#pragma unroll
    for (int i2 = 0; i2 < 8; ++i2) {
      int k0 = kb + i2, k1 = 16 + kb + i2;
      aq[i2]   = (k0 < kC) ? bf(qr[k0]) : bfu(0);
      aq[8+i2] = (k1 < kC) ? bf(qr[k1]) : bfu(0);
    }
    const float* qpr = qp + (size_t)(b*kNn + i0 + r) * (kH*kPQ*3);
#pragma unroll
    for (int i2 = 0; i2 < 8; ++i2) {
#pragma unroll
      for (int s = 0; s < 2; ++s) {
        int K = s*16 + kb + i2;
        __bf16 v = bfu(0);
        if (K < 3*kPQ) {
          int p = K / 3, x = K - 3*p;
          v = bf(qpr[(x*kH + h)*kPQ + p] + tI[r][x]);
        }
        ax[s*8 + i2] = v;
      }
    }
  }
#endif

  // ------- phase 1: logit tiles -------
  for (int jt = wave; jt < kNn/16; jt += 4) {
    const int j0 = jt * 16;
    const int jl = j0 + r;                 // this lane's j column
    const float mJ = mask[b*kNn + jl];
    float tJ[3];
#pragma unroll
    for (int d = 0; d < 3; ++d) tJ[d] = x_t[(b*kNn + jl)*3 + d] * mJ;
    const float sqkJ = sqk[(b*kH + h)*kNn + jl];

    v8f dqk = {}, dx = {};
#ifdef HAVE_WMMA_F32X4
    {
      const float* kvr = kv + (size_t)(b*kNn + jl) * (kH*2*kC) + h*2*kC;
#pragma unroll
      for (int s = 0; s < 4; ++s) {
        v2f a2 = { aqf[2*s], aqf[2*s+1] };
        v2f b2 = { kvr[4*s + 2*half + 0], kvr[4*s + 2*half + 1] };
        dqk = wmma_f32x4(a2, b2, dqk);
      }
      const float* kpr = kvp + (size_t)(b*kNn + jl) * (kH*(kPQ+kPV)*3);
#pragma unroll
      for (int s = 0; s < 6; ++s) {
        float bv[2];
#pragma unroll
        for (int t = 0; t < 2; ++t) {
          int K = 4*s + 2*half + t;
          int p = K / 3, x = K - 3*p;
          bv[t] = kpr[(x*kH + h)*(kPQ+kPV) + p] + tJ[x];
        }
        v2f a2 = { axf[2*s], axf[2*s+1] };
        v2f b2 = { bv[0], bv[1] };
        dx = wmma_f32x4(a2, b2, dx);
      }
    }
#else
    {
      v16bf bk, bx;
      const float* kvr = kv + (size_t)(b*kNn + jl) * (kH*2*kC) + h*2*kC;
#pragma unroll
      for (int m = 0; m < 16; ++m) {
        int K = kb + m + (m & 8);
        bk[m] = (K < kC) ? bf(kvr[K]) : bfu(0);
      }
      const float* kpr = kvp + (size_t)(b*kNn + jl) * (kH*(kPQ+kPV)*3);
#pragma unroll
      for (int m = 0; m < 16; ++m) {
        int K = kb + m + (m & 8);
        __bf16 v = bfu(0);
        if (K < 3*kPQ) {
          int p = K / 3, x = K - 3*p;
          v = bf(kpr[(x*kH + h)*(kPQ+kPV) + p] + tJ[x]);
        }
        bx[m] = v;
      }
      dqk = wmma_bf16(aq, bk, dqk);
      dx  = wmma_bf16(ax, bx, dx);
    }
#endif
#pragma unroll
    for (int rr = 0; rr < 8; ++rr) {
      int il = rbase + rr;
      float d2 = sq16[il] + sqkJ - 2.f * dx[rr];
      float lg = dqk[rr] * kWL
               + kS3 * biasw[((size_t)(b*kNn + i0 + il) * kNn + jl) * kH + h]
               - 0.5f * hw * d2
               + kINF * (mI[il] * mJ - 1.f);
      S[il][jl] = lg;
    }
  }
  __syncthreads();

  // ------- softmax over j (rows of S) -------
  {
    const int row = tid >> 3, sub = tid & 7;
    float mx = -3.4e38f;
    for (int j = sub; j < kNn; j += 8) mx = fmaxf(mx, S[row][j]);
    red[row][sub] = mx;
    __syncthreads();
    if (sub == 0) {
      float m2 = red[row][0];
#pragma unroll
      for (int s = 1; s < 8; ++s) m2 = fmaxf(m2, red[row][s]);
      rowst[row] = m2;
    }
    __syncthreads();
    const float rm = rowst[row];
    float sm = 0.f;
    for (int j = sub; j < kNn; j += 8) {
      float e = expf(S[row][j] - rm);
      S[row][j] = e;
      sm += e;
    }
    __syncthreads();
    red[row][sub] = sm;
    __syncthreads();
    if (sub == 0) {
      float s2 = red[row][0];
#pragma unroll
      for (int s = 1; s < 8; ++s) s2 += red[row][s];
      rowst[row] = s2;
    }
    __syncthreads();
    const float inv = 1.f / rowst[row];
    for (int j = sub; j < kNn; j += 8) {
      u16 pb = bfbits(S[row][j] * inv);
      Pl[row*kNn + j] = pb;
      Pg[((size_t)(b*kH + h) * kNn + (i0 + row)) * kNn + j] = pb;
    }
  }
  __syncthreads();

  // ------- phase 2: o (wave 0) / o_pt (waves 1..3), K = 512 -------
  {
    v8f acc = {};
    for (int k0 = 0; k0 < kNn; k0 += 32) {
      v16bf a, bm;
      const u16* prow = Pl + r*kNn + k0;
#pragma unroll
      for (int i2 = 0; i2 < 8; ++i2) {
        a[i2]   = bfu(prow[kb + i2]);
        a[8+i2] = bfu(prow[16 + kb + i2]);
      }
      if (wave == 0) {
#pragma unroll
        for (int m = 0; m < 16; ++m) {
          int j = k0 + kb + m + (m & 8);
          bm[m] = bf(kv[(size_t)(b*kNn + j) * (kH*2*kC) + h*2*kC + kC + r]);
        }
      } else {
        const int cg = (wave-1)*16 + r;
        const int p = cg / 3, x = cg - 3*p;
        const bool valid = (cg < 3*kPV);
#pragma unroll
        for (int m = 0; m < 16; ++m) {
          int j = k0 + kb + m + (m & 8);
          __bf16 v = bfu(0);
          if (valid) {
            float mj = mask[b*kNn + j];
            float tj = x_t[(b*kNn + j)*3 + x] * mj;
            v = bf(kvp[(size_t)(b*kNn + j) * (kH*(kPQ+kPV)*3)
                       + (x*kH + h)*(kPQ+kPV) + kPQ + p] + tj);
          }
          bm[m] = v;
        }
      }
      acc = wmma_bf16(a, bm, acc);
    }
    if (wave == 0) {
#pragma unroll
      for (int rr = 0; rr < 8; ++rr)
        feats[(size_t)(b*kNn + i0 + rbase + rr) * kCat + h*kC + r] = acc[rr];
    } else {
      const int cg = (wave-1)*16 + r;
#pragma unroll
      for (int rr = 0; rr < 8; ++rr)
        Opt[rbase + rr][cg] = acc[rr];
    }
  }
  __syncthreads();

  // ------- o_pt epilogue: subtract t, write xyz (split-coord layout) + norm
  for (int idx = tid; idx < 16*kPV; idx += 128) {
    int il = idx / kPV, p = idx - il*kPV;
    float x = Opt[il][p*3+0] - tI[il][0];
    float y = Opt[il][p*3+1] - tI[il][1];
    float z = Opt[il][p*3+2] - tI[il][2];
    size_t rowb = (size_t)(b*kNn + i0 + il) * kCat;
    feats[rowb + kH*kC + 0*(kH*kPV) + h*kPV + p] = x;
    feats[rowb + kH*kC + 1*(kH*kPV) + h*kPV + p] = y;
    feats[rowb + kH*kC + 2*(kH*kPV) + h*kPV + p] = z;
    feats[rowb + kH*kC + 3*(kH*kPV) + h*kPV + p] = sqrtf(x*x + y*y + z*z + kEPS);
  }
}

// ==========================================================================
// K5: o_pair per (b,i):  [16(h pad) x 512] @ pair_rep[b,i] (512 x 128)
// 256 threads = 8 waves = 8 n-tiles of 16 z-columns.
// pair_rep (dominant HBM stream) staged via async global->LDS, double buffered.
// ==========================================================================
__global__ __launch_bounds__(256)
void k_opair(const u16* __restrict__ Pg, const float* __restrict__ pr,
             float* __restrict__ feats) {
  const int bi = blockIdx.x;
  const int b = bi / kNn, i = bi - b*kNn;
  const int tid = threadIdx.x, lane = tid & 31, wave = tid >> 5;
  const int r = lane & 15, half = lane >> 4;
  const int kb = half ? 8 : 0, ks = half ? 16 : 0;

  __shared__ u16   Pls[16 * kNn];       // rows = heads (8 valid, 8 zero)  16 KB
  __shared__ float Bt[2][32 * kCZ];     // staged pair_rep chunks          32 KB

  for (int idx = tid; idx < 16*kNn; idx += 256) {
    int hh = idx / kNn, j = idx - hh*kNn;
    Pls[idx] = (hh < kH) ? Pg[((size_t)(b*kH + hh) * kNn + i) * kNn + j] : (u16)0;
  }

  const float* prb = pr + (size_t)(b*kNn + i) * kNn * kCZ;

  auto stage = [&](int buf, int kt) {
    const float* g = prb + (size_t)kt * 32 * kCZ;
#pragma unroll
    for (int rr = 0; rr < 4; ++rr) {
      int idx4 = rr * 256 + tid;        // float4 index inside 32x128 tile
      stage_async(g + idx4*4, &Bt[buf][idx4*4]);
    }
  };

  stage(0, 0);
  wait_async();
  __syncthreads();

  v8f acc = {};
  const int col = wave*16 + r;          // z column
  for (int kt = 0; kt < kNn/32; ++kt) {
    if (kt + 1 < kNn/32) stage((kt+1) & 1, kt+1);
    const float* Bc = Bt[kt & 1];
    const u16* prow = Pls + r*kNn + kt*32;
    v16bf a, bm;
#pragma unroll
    for (int i2 = 0; i2 < 8; ++i2) {
      a[i2]   = bfu(prow[kb + i2]);
      a[8+i2] = bfu(prow[16 + kb + i2]);
    }
#pragma unroll
    for (int m = 0; m < 16; ++m)
      bm[m] = bf(Bc[(kb + m + (m & 8))*kCZ + col]);
    acc = wmma_bf16(a, bm, acc);
    wait_async();
    __syncthreads();
  }
  if (half == 0) {                      // rows 0..7 == valid heads
#pragma unroll
    for (int rr = 0; rr < 8; ++rr)
      feats[(size_t)(b*kNn + i) * kCat + (kH*kC + kH*kPV*4) + rr*kCZ + col] = acc[rr];
  }
}

// ==========================================================================
// K6: out = (seqs + feats @ Wout + bout) * mask
// ==========================================================================
__global__ __launch_bounds__(32)
void k_out(const float* __restrict__ feats, const float* __restrict__ Wout,
           const float* __restrict__ bout, const float* __restrict__ seqs,
           const float* __restrict__ mask, float* __restrict__ out) {
  const int mt = blockIdx.x, nt = blockIdx.y;
  const int lane = threadIdx.x;
  const int r = lane & 15, half = lane >> 4, rbase = half*8;
  const int kb = half ? 8 : 0, ks = half ? 16 : 0;
  const float* Ar = feats + (size_t)(mt*16 + r) * kCat;
  const int col = nt*16 + r;
  v8f acc = {};
  for (int k0 = 0; k0 < kCat; k0 += 32) {
    v16bf a, bm;
#pragma unroll
    for (int i2 = 0; i2 < 8; ++i2) {
      a[i2]   = bf(Ar[k0 + kb + i2]);
      a[8+i2] = bf(Ar[k0 + 16 + kb + i2]);
    }
#pragma unroll
    for (int m = 0; m < 16; ++m)
      bm[m] = bf(Wout[(size_t)(k0 + kb + m + (m & 8)) * kCS + col]);
    acc = wmma_bf16(a, bm, acc);
  }
  const float bv = bout[col];
#pragma unroll
  for (int rr = 0; rr < 8; ++rr) {
    int row = mt*16 + rbase + rr;       // row = b*N + n; mask is (B,N) flat
    float mval = mask[row];
    out[(size_t)row * kCS + col] = (seqs[(size_t)row * kCS + col] + acc[rr] + bv) * mval;
  }
}

// ==========================================================================
extern "C" void kernel_launch(void* const* d_in, const int* in_sizes, int n_in,
                              void* d_out, int out_size, void* d_ws, size_t ws_size,
                              hipStream_t stream) {
  const float* seqs = (const float*)d_in[0];
  const float* pr   = (const float*)d_in[1];
  const float* x_t  = (const float*)d_in[2];
  const float* mask = (const float*)d_in[3];
  const float* Wq   = (const float*)d_in[4];
  const float* bq   = (const float*)d_in[5];
  const float* Wkv  = (const float*)d_in[6];
  const float* bkv  = (const float*)d_in[7];
  const float* Wqp  = (const float*)d_in[8];
  const float* bqp  = (const float*)d_in[9];
  const float* Wkvp = (const float*)d_in[10];
  const float* bkvp = (const float*)d_in[11];
  const float* Wb   = (const float*)d_in[12];
  const float* bb   = (const float*)d_in[13];
  const float* hwv  = (const float*)d_in[14];
  const float* Wout = (const float*)d_in[15];
  const float* bout = (const float*)d_in[16];
  float* out = (float*)d_out;

  // workspace carve-up (floats); total ~27.5 MB
  float* ws    = (float*)d_ws;
  float* q     = ws;                       // 1024*128
  float* kv    = q     + 131072;           // 1024*256
  float* qp    = kv    + 262144;           // 1024*192
  float* kvp   = qp    + 196608;           // 1024*480
  float* biasw = kvp   + 491520;           // 2*512*512*8
  float* sqq   = biasw + 2097152;          // 2*8*512
  float* sqk   = sqq   + 8192;
  float* feats = sqk   + 8192;             // 1024*1536
  u16*   Pg    = (u16*)(feats + 1572864);  // 2*8*512*512 bf16 bits

  const int M = kBn * kNn;                 // 1024 tokens

  k_proj<<<dim3(M/16, (kH*kC)/16),            32, 0, stream>>>(seqs, Wq,   bq,   q,   kCS, kH*kC);
  k_proj<<<dim3(M/16, (kH*2*kC)/16),          32, 0, stream>>>(seqs, Wkv,  bkv,  kv,  kCS, kH*2*kC);
  k_proj<<<dim3(M/16, (kH*kPQ*3)/16),         32, 0, stream>>>(seqs, Wqp,  bqp,  qp,  kCS, kH*kPQ*3);
  k_proj<<<dim3(M/16, (kH*(kPQ+kPV)*3)/16),   32, 0, stream>>>(seqs, Wkvp, bkvp, kvp, kCS, kH*(kPQ+kPV)*3);
  k_sq  <<<(kBn*kH*kNn + 255)/256, 256, 0, stream>>>(qp, kvp, x_t, mask, sqq, sqk);
  k_bias<<<(kBn*kNn*kNn)/64, 128, 0, stream>>>(pr, Wb, bb, biasw);
  k_attn<<<dim3(kNn/16, kH, kBn), 128, 0, stream>>>(q, kv, qp, kvp, x_t, mask,
                                                    biasw, sqq, sqk, hwv, Pg, feats);
  k_opair<<<kBn*kNn, 256, 0, stream>>>(Pg, pr, feats);
  k_out<<<dim3(M/16, kCS/16), 32, 0, stream>>>(feats, Wout, bout, seqs, mask, out);

  (void)in_sizes; (void)n_in; (void)out_size; (void)ws_size;
}
